// Period_Conv_Attention_35150012350773
// MI455X (gfx1250) — matrix operand order repaired and checked by reading, not compiled, not run on hardware
//
#include <hip/hip_runtime.h>

// MI455X / gfx1250, wave32. Matrix ops via CDNA5 WMMA intrinsics:
//  - projections (fp32):  v_wmma_f32_16x16x4_f32
//  - attention scores:    v_wmma_f32_16x16x32_bf16
// Workspace layout (bf16): wq[16][512][4096] | wk[16][512][4096] | wv[16][512][2048]
// => 160 MiB required in d_ws.

typedef __attribute__((ext_vector_type(2)))  float  v2f;
typedef __attribute__((ext_vector_type(8)))  float  v8f;
typedef __attribute__((ext_vector_type(16))) __bf16 v16bf;

#define DIM   512
#define LTOT  4096
#define NB    16      // b * nvar
#define NH    32
#define SROW  65      // padded LDS row stride for 64x64 scores

__device__ __forceinline__ unsigned short f2bf(float f) {
  unsigned int u = __float_as_uint(f);
  u += 0x7FFFu + ((u >> 16) & 1u);     // round-to-nearest-even
  return (unsigned short)(u >> 16);
}
__device__ __forceinline__ float bf2f(unsigned short h) {
  return __uint_as_float(((unsigned int)h) << 16);
}

// ---------------------------------------------------------------------------
// Kernel 1: Y[m, l] = sum_c W[m,c] * X[b,c,l] + bias[m], stored as bf16.
// Grid: (Ncols/64, 512/128, 16 batches), block = 256 threads = 8 waves.
// Each wave: one 16(M) x 64(N) tile, fp32 WMMA 16x16x4, K-loop of 128.
// ---------------------------------------------------------------------------
__global__ __launch_bounds__(256) void proj_bf16_kernel(
    const float* __restrict__ W, const float* __restrict__ bias,
    const float* __restrict__ X, unsigned short* __restrict__ dst,
    int dstStride)
{
  const int lane = threadIdx.x & 31;
  const int wave = threadIdx.x >> 5;
  const int mr   = lane & 15;     // M index within tile (A/B/D lane row/col)
  const int kh   = lane >> 4;     // K half-select per f32 WMMA layout

  const int m0 = blockIdx.y * 128 + wave * 16;
  const int n0 = blockIdx.x * 64;
  const float* Xb = X + (size_t)blockIdx.z * DIM * LTOT;
  unsigned short* Db = dst + (size_t)blockIdx.z * DIM * dstStride;

  v8f acc0 = {}, acc1 = {}, acc2 = {}, acc3 = {};
  // A (16x4 f32): lane holds {W[m, k+2*kh], W[m, k+2*kh+1]}
  const float* Arow = W  + (size_t)(m0 + mr) * DIM + 2 * kh;
  // B (4x16 f32): lane holds {X[k+2*kh, n], X[k+2*kh+1, n]}
  const float* Bcol = Xb + (size_t)(2 * kh) * LTOT + n0 + mr;

  for (int k = 0; k < DIM; k += 4) {
    v2f a = *(const v2f*)(Arow + k);
    const float* bp = Bcol + (size_t)k * LTOT;
    v2f b0 = { bp[ 0], bp[ 0 + LTOT] };
    v2f b1 = { bp[16], bp[16 + LTOT] };
    v2f b2 = { bp[32], bp[32 + LTOT] };
    v2f b3 = { bp[48], bp[48 + LTOT] };
    acc0 = __builtin_amdgcn_wmma_f32_16x16x4_f32(false, a, false, b0, (short)0, acc0, false, false);
    acc1 = __builtin_amdgcn_wmma_f32_16x16x4_f32(false, a, false, b1, (short)0, acc1, false, false);
    acc2 = __builtin_amdgcn_wmma_f32_16x16x4_f32(false, a, false, b2, (short)0, acc2, false, false);
    acc3 = __builtin_amdgcn_wmma_f32_16x16x4_f32(false, a, false, b3, (short)0, acc3, false, false);
  }

  // D layout: VGPR r -> M = kh*8 + r, N = mr (per 16x16 f32 C/D table)
  #pragma unroll
  for (int r = 0; r < 8; ++r) {
    const int m = m0 + kh * 8 + r;
    const float bb = bias[m];
    unsigned short* drow = Db + (size_t)m * dstStride + n0 + mr;
    drow[ 0] = f2bf(acc0[r] + bb);
    drow[16] = f2bf(acc1[r] + bb);
    drow[32] = f2bf(acc2[r] + bb);
    drow[48] = f2bf(acc3[r] + bb);
  }
}

// ---------------------------------------------------------------------------
// Kernel 2: per (b,d) pair: scores = (wq_rs)^T wk_rs * scale (bf16 WMMA),
// softmax -> attn (full 64x64 written out), then out32[p] = attn[0,:] . wv[p,:],
// final 32x32 Wo projection. One wave per pair, 2 waves per workgroup.
// ---------------------------------------------------------------------------
__global__ __launch_bounds__(64) void attn_softmax_kernel(
    const unsigned short* __restrict__ wq,
    const unsigned short* __restrict__ wk,
    const unsigned short* __restrict__ wv2,
    const float* __restrict__ Wo, const float* __restrict__ bo,
    float* __restrict__ outp, float* __restrict__ attn)
{
  __shared__ float ldsbuf[2][64 * SROW + 64 + 64 + 32];
  const int lane = threadIdx.x & 31;
  const int wave = threadIdx.x >> 5;
  float* sc  = ldsbuf[wave];          // 64 x 64 scores, row stride SROW
  float* mx  = sc + 64 * SROW;        // per-row max
  float* inv = mx + 64;               // per-row 1/sum
  float* o32 = inv + 64;              // surviving 32 outputs

  const int pair = blockIdx.x * 2 + wave;      // pair = b*512 + d
  const unsigned short* q  = wq  + (size_t)pair * 4096;
  const unsigned short* kk = wk  + (size_t)pair * 4096;
  const unsigned short* vv = wv2 + (size_t)pair * 2048;

  const int mr = lane & 15;
  const int hh = lane >> 4;

  typedef union { unsigned short u[16]; v16bf v; } fragU;

  // A fragments: A[m=i, K=p] = wq[p*64 + i]; 16-bit A 16x32 lane layout:
  // lane<16: K = t + (t>=8 ? 8:0); lane>=16: +8  ==>  K = hh*8 + ((t>>3)<<4) + (t&7)
  fragU A[8];
  #pragma unroll
  for (int it = 0; it < 4; ++it) {
    const int i = it * 16 + mr;
    #pragma unroll
    for (int t = 0; t < 16; ++t) {
      const int K = hh * 8 + ((t >> 3) << 4) + (t & 7);
      A[it * 2 + 0].u[t] = q[(K)      * 64 + i];
      A[it * 2 + 1].u[t] = q[(K + 32) * 64 + i];
    }
  }

  const float scale = 0.04419417382415922f;    // 512^-0.5
  for (int jt = 0; jt < 4; ++jt) {
    // B fragments: B[K=p, n=j] = wk[p*64 + j]; lane layout K = hh*8 + ((t>>3)<<4) + (t&7)
    fragU blo, bhi;
    const int j = jt * 16 + mr;
    #pragma unroll
    for (int t = 0; t < 16; ++t) {
      const int K = hh * 8 + ((t >> 3) << 4) + (t & 7);
      blo.u[t] = kk[(K)      * 64 + j];
      bhi.u[t] = kk[(K + 32) * 64 + j];
    }
    #pragma unroll
    for (int it = 0; it < 4; ++it) {
      v8f acc = {};
      acc = __builtin_amdgcn_wmma_f32_16x16x32_bf16(false, A[it*2+0].v, false, blo.v, (short)0, acc, false, false);
      acc = __builtin_amdgcn_wmma_f32_16x16x32_bf16(false, A[it*2+1].v, false, bhi.v, (short)0, acc, false, false);
      #pragma unroll
      for (int r = 0; r < 8; ++r) {
        const int ii = it * 16 + hh * 8 + r;
        sc[ii * SROW + (jt * 16 + mr)] = acc[r] * scale;
      }
    }
  }

  // Row softmax stats: each lane owns rows lane*2 and lane*2+1.
  #pragma unroll
  for (int rr = 0; rr < 2; ++rr) {
    const int row = lane * 2 + rr;
    float m = -3.0e38f;
    for (int jj = 0; jj < 64; ++jj) m = fmaxf(m, sc[row * SROW + jj]);
    float s = 0.0f;
    for (int jj = 0; jj < 64; ++jj) s += __expf(sc[row * SROW + jj] - m);
    mx[row]  = m;
    inv[row] = 1.0f / s;
  }

  // Coalesced attn write-out (float4 per lane per iteration).
  float* arow = attn + (size_t)pair * 4096;
  for (int t = 0; t < 32; ++t) {
    const int flat = (t * 32 + lane) * 4;
    const int row = flat >> 6;
    const int col = flat & 63;
    const float m = mx[row], iv = inv[row];
    float4 o;
    o.x = __expf(sc[row * SROW + col + 0] - m) * iv;
    o.y = __expf(sc[row * SROW + col + 1] - m) * iv;
    o.z = __expf(sc[row * SROW + col + 2] - m) * iv;
    o.w = __expf(sc[row * SROW + col + 3] - m) * iv;
    *(float4*)(arow + flat) = o;
  }

  // Only out[i=0, p=0..31] survives the slice: 32 dots of length 64.
  {
    const float m0v = mx[0], i0v = inv[0];
    float accv = 0.0f;
    for (int jj = 0; jj < 64; ++jj) {
      const float a0 = __expf(sc[jj] - m0v) * i0v;     // attn row 0
      accv += a0 * bf2f(vv[lane * 64 + jj]);           // wv[d, p*64 + j], p=lane
    }
    o32[lane] = accv;
  }

  // Output projection: out[o] = bo[o] + sum_h Wo[o,h] * o32[h]
  {
    float r = bo[lane];
    for (int h = 0; h < NH; ++h) r += Wo[lane * NH + h] * o32[h];
    outp[(size_t)pair * NH + lane] = r;
  }
}

// ---------------------------------------------------------------------------
extern "C" void kernel_launch(void* const* d_in, const int* in_sizes, int n_in,
                              void* d_out, int out_size, void* d_ws, size_t ws_size,
                              hipStream_t stream) {
  (void)in_sizes; (void)n_in; (void)out_size; (void)ws_size;
  const float* Q  = (const float*)d_in[0];
  const float* K  = (const float*)d_in[1];
  const float* V  = (const float*)d_in[2];
  const float* Wq = (const float*)d_in[3];
  const float* bq = (const float*)d_in[4];
  const float* Wk = (const float*)d_in[5];
  const float* bk = (const float*)d_in[6];
  const float* Wv = (const float*)d_in[7];
  const float* bv = (const float*)d_in[8];
  const float* Wo = (const float*)d_in[9];
  const float* bo = (const float*)d_in[10];

  unsigned short* wq  = (unsigned short*)d_ws;
  unsigned short* wk  = wq + (size_t)NB * DIM * LTOT;
  unsigned short* wv2 = wk + (size_t)NB * DIM * LTOT;   // only l < 2048 stored

  float* outp = (float*)d_out;                          // [16*512, 32]
  float* attn = outp + (size_t)NB * DIM * NH;           // [16*512, 64, 64]

  proj_bf16_kernel<<<dim3(64, 4, 16), dim3(256), 0, stream>>>(Wq, bq, Q, wq,  LTOT);
  proj_bf16_kernel<<<dim3(64, 4, 16), dim3(256), 0, stream>>>(Wk, bk, K, wk,  LTOT);
  proj_bf16_kernel<<<dim3(32, 4, 16), dim3(256), 0, stream>>>(Wv, bv, V, wv2, 2048);
  attn_softmax_kernel<<<dim3(4096), dim3(64), 0, stream>>>(wq, wk, wv2, Wo, bo, outp, attn);
}
